// GATEncoder_60808146977390
// MI455X (gfx1250) — compile-verified, matrix operand order checked by reading, not run on hardware
//
#include <hip/hip_runtime.h>

typedef __attribute__((ext_vector_type(16))) _Float16 v16h;
typedef __attribute__((ext_vector_type(8)))  _Float16 v8h;
typedef __attribute__((ext_vector_type(8)))  float    v8f;

#define NEG_SLOPE 0.2f
#define HEADS 4
#define CH 64
#define FDIM 256   // HEADS*CH

// ---------------- utility: fill ----------------
__global__ void fill_kernel(float* __restrict__ p, long long n, float v) {
  long long i = (long long)blockIdx.x * blockDim.x + threadIdx.x;
  if (i < n) p[i] = v;
}

// ---------------- f32 -> f16 row-major copy ----------------
__global__ void cvt_f16_kernel(const float* __restrict__ src,
                               _Float16* __restrict__ dst, long long n) {
  long long i = (long long)blockIdx.x * blockDim.x + threadIdx.x;
  if (i < n) dst[i] = (_Float16)src[i];
}

// ---------------- f32 W[K,N] -> f16 Bt[N,K] (transpose) ----------------
__global__ void cvt_f16_T_kernel(const float* __restrict__ W,
                                 _Float16* __restrict__ Bt, int K, int N) {
  int i = blockIdx.x * blockDim.x + threadIdx.x;   // (n, k)
  if (i >= K * N) return;
  int n = i / K, k = i - n * K;
  Bt[i] = (_Float16)W[(long long)k * N + n];
}

// ---------------- WMMA GEMM (f16 inputs, f32 accumulate) ----------------
// C[M,N] = Ah[M,K] @ Bt[N,K]^T, f16 row-major A, f16 transposed B, f32 C.
// Requires M%16==0, K%32==0, N%128==0. One wave computes one 16x16 C tile.
// Block = 256 threads = 8 waves; gridDim.x = N/128, gridDim.y = M/16.
__global__ __launch_bounds__(256)
void gemm_wmma_kernel(const _Float16* __restrict__ Ah,
                      const _Float16* __restrict__ Bt,
                      float* __restrict__ C,
                      int M, int K, int N) {
  const int lane = threadIdx.x & 31;
  const int wave = threadIdx.x >> 5;
  const int n0   = (blockIdx.x * 8 + wave) * 16;
  const int m0   = blockIdx.y * 16;
  const int half = (lane < 16) ? 0 : 8;   // K-offset / M-offset for upper half-wave
  const int l16  = lane & 15;
  const _Float16* ap = Ah + (long long)(m0 + l16) * K + half; // A row this lane feeds
  const _Float16* bp = Bt + (long long)(n0 + l16) * K + half; // B col this lane feeds
  const int bcol = n0 + l16;

  v8f acc = {};
  for (int k0 = 0; k0 < K; k0 += 32) {
    // Fragment = halves [kb..kb+7] and [kb+16..kb+23]; both 16B aligned.
    const v8h alo = *(const v8h*)(ap + k0);
    const v8h ahi = *(const v8h*)(ap + k0 + 16);
    const v8h blo = *(const v8h*)(bp + k0);
    const v8h bhi = *(const v8h*)(bp + k0 + 16);
    const v16h a = __builtin_shufflevector(alo, ahi, 0, 1, 2, 3, 4, 5, 6, 7,
                                           8, 9, 10, 11, 12, 13, 14, 15);
    const v16h b = __builtin_shufflevector(blo, bhi, 0, 1, 2, 3, 4, 5, 6, 7,
                                           8, 9, 10, 11, 12, 13, 14, 15);
    acc = __builtin_amdgcn_wmma_f32_16x16x32_f16(
        /*neg_a=*/false, a, /*neg_b=*/false, b,
        /*c_mod=*/(short)0, acc, /*reuse_a=*/false, /*reuse_b=*/false);
  }
  // C tile: VGPR v -> row m0 + v + (lane<16 ? 0 : 8), col = bcol
#pragma unroll
  for (int v = 0; v < 8; ++v) {
    C[(long long)(m0 + half + v) * N + bcol] = acc[v];
  }
}

// ---------------- attention projections ----------------
__global__ void attn_proj_kernel(const float* __restrict__ hmat,
                                 const float* __restrict__ att_src,
                                 const float* __restrict__ att_dst,
                                 float* __restrict__ asrc,
                                 float* __restrict__ adst,
                                 int Nn) {
  int i = blockIdx.x * blockDim.x + threadIdx.x;   // (node, head)
  if (i >= Nn * HEADS) return;
  int hd   = i & (HEADS - 1);
  int node = i >> 2;
  const float* hp = hmat + (long long)node * FDIM + hd * CH;
  const float* as = att_src + hd * CH;
  const float* ad = att_dst + hd * CH;
  float s = 0.f, d = 0.f;
#pragma unroll 8
  for (int c = 0; c < CH; ++c) {
    float v = hp[c];
    s += v * as[c];
    d += v * ad[c];
  }
  asrc[i] = s;
  adst[i] = d;
}

// ---------------- float atomic max (bit-punning trick) ----------------
__device__ __forceinline__ void atomicMaxF(float* addr, float val) {
  if (val >= 0.f)
    atomicMax((int*)addr, __float_as_int(val));
  else
    atomicMin((unsigned int*)addr, __float_as_uint(val));
}

__device__ __forceinline__ void edge_nodes(int e, int E,
                                           const long long* __restrict__ esrc,
                                           const long long* __restrict__ edst,
                                           long long& s, long long& d) {
  if (e < E) { s = esrc[e]; d = edst[e]; }
  else       { s = d = (long long)(e - E); }   // appended self-loops
}

// ---------------- edge pass 1: segment max ----------------
__global__ void edge_max_kernel(const float* __restrict__ asrc,
                                const float* __restrict__ adst,
                                const long long* __restrict__ esrc,
                                const long long* __restrict__ edst,
                                float* __restrict__ nmax,
                                int E, int EA) {
  int i = blockIdx.x * blockDim.x + threadIdx.x;   // (edge, head)
  if (i >= EA * HEADS) return;
  int e = i >> 2, hd = i & (HEADS - 1);
  long long s, d;
  edge_nodes(e, E, esrc, edst, s, d);
  float v = asrc[s * HEADS + hd] + adst[d * HEADS + hd];
  v = (v > 0.f) ? v : v * NEG_SLOPE;               // leaky_relu
  atomicMaxF(&nmax[d * HEADS + hd], v);
}

// ---------------- edge pass 2: exp + segment sum ----------------
__global__ void edge_exp_kernel(const float* __restrict__ asrc,
                                const float* __restrict__ adst,
                                const long long* __restrict__ esrc,
                                const long long* __restrict__ edst,
                                const float* __restrict__ nmax,
                                float* __restrict__ nsum,
                                float* __restrict__ eexp,
                                int E, int EA) {
  int i = blockIdx.x * blockDim.x + threadIdx.x;   // (edge, head)
  if (i >= EA * HEADS) return;
  int e = i >> 2, hd = i & (HEADS - 1);
  long long s, d;
  edge_nodes(e, E, esrc, edst, s, d);
  float v = asrc[s * HEADS + hd] + adst[d * HEADS + hd];
  v = (v > 0.f) ? v : v * NEG_SLOPE;
  float ex = expf(v - nmax[d * HEADS + hd]);
  eexp[i] = ex;
  atomicAdd(&nsum[d * HEADS + hd], ex);
}

// ---------------- edge pass 3: alpha-weighted message scatter ----------------
// One thread per (edge, float4-chunk of 256 channels): 64 threads/edge.
__global__ void edge_msg_kernel(const float* __restrict__ hmat,
                                const long long* __restrict__ esrc,
                                const long long* __restrict__ edst,
                                const float* __restrict__ nsum,
                                const float* __restrict__ eexp,
                                float* __restrict__ acc,
                                int E, int EA) {
  long long i = (long long)blockIdx.x * blockDim.x + threadIdx.x;
  if (i >= (long long)EA * 64) return;
  int e  = (int)(i >> 6);
  int c4 = (int)(i & 63);       // which float4 within 256-channel row
  int hd = c4 >> 4;             // 16 float4s per head
  long long s, d;
  edge_nodes(e, E, esrc, edst, s, d);
  float alpha = eexp[(long long)e * HEADS + hd] /
                (nsum[d * HEADS + hd] + 1e-16f);
  const float4 hv = ((const float4*)(hmat + (long long)s * FDIM))[c4];
  float* ap = acc + (long long)d * FDIM + c4 * 4;
  atomicAdd(ap + 0, hv.x * alpha);
  atomicAdd(ap + 1, hv.y * alpha);
  atomicAdd(ap + 2, hv.z * alpha);
  atomicAdd(ap + 3, hv.w * alpha);
}

// ---------------- bias + ELU (in place) ----------------
__global__ void bias_elu_kernel(float* __restrict__ acc,
                                const float* __restrict__ bias,
                                long long n) {
  long long i = (long long)blockIdx.x * blockDim.x + threadIdx.x;
  if (i >= n) return;
  float x = acc[i] + bias[(int)(i & (FDIM - 1))];
  acc[i] = (x > 0.f) ? x : (expf(x) - 1.f);
}

// ---------------- head mean + bias -> output ----------------
__global__ void final_mean_kernel(const float* __restrict__ acc,
                                  const float* __restrict__ b2,
                                  float* __restrict__ out,
                                  int Nn) {
  int i = blockIdx.x * blockDim.x + threadIdx.x;   // (node, channel)
  if (i >= Nn * CH) return;
  int node = i >> 6, c = i & 63;
  const float* ap = acc + (long long)node * FDIM + c;
  out[i] = (ap[0] + ap[64] + ap[128] + ap[192]) * 0.25f + b2[c];
}

// ---------------- host orchestration ----------------
static inline int cdiv(long long a, int b) { return (int)((a + b - 1) / b); }

extern "C" void kernel_launch(void* const* d_in, const int* in_sizes, int n_in,
                              void* d_out, int out_size, void* d_ws, size_t ws_size,
                              hipStream_t stream) {
  (void)n_in; (void)out_size; (void)ws_size;
  const float*     x    = (const float*)d_in[0];
  const long long* eidx = (const long long*)d_in[1];   // int64 [2, E]
  const float*     W1   = (const float*)d_in[2];
  const float*     as1  = (const float*)d_in[3];
  const float*     ad1  = (const float*)d_in[4];
  const float*     b1   = (const float*)d_in[5];
  const float*     W2   = (const float*)d_in[6];
  const float*     as2  = (const float*)d_in[7];
  const float*     ad2  = (const float*)d_in[8];
  const float*     b2   = (const float*)d_in[9];

  const int K1 = 128;
  const int Nn = in_sizes[0] / K1;        // 50000
  const int E  = in_sizes[1] / 2;         // 800000
  const int EA = E + Nn;                  // with self-loops
  const int K2 = in_sizes[6] / FDIM;      // 256
  const long long* esrc = eidx;
  const long long* edst = eidx + E;

  // workspace layout (all offsets are multiples of 16 bytes)
  float* ws   = (float*)d_ws;
  float* hbuf = ws;                              // [Nn, 256] f32  h1 then h2
  float* acc  = hbuf + (size_t)Nn * FDIM;        // [Nn, 256] f32  acc1 -> hx -> acc2
  float* eexp = acc  + (size_t)Nn * FDIM;        // [EA, 4]   f32
  float* asrc = eexp + (size_t)EA * HEADS;       // [Nn, 4]   f32
  float* adst = asrc + (size_t)Nn * HEADS;       // [Nn, 4]   f32
  float* nmax = adst + (size_t)Nn * HEADS;       // [Nn, 4]   f32
  float* nsum = nmax + (size_t)Nn * HEADS;       // [Nn, 4]   f32
  _Float16* ah16 = (_Float16*)(nsum + (size_t)Nn * HEADS); // [Nn, 256] f16 (A operand)
  _Float16* bt16 = ah16 + (size_t)Nn * FDIM;               // [256, 256] f16 (W^T operand)

  const int B = 256;
  dim3 gemm_grid(FDIM / 128, Nn / 16);
  const float NEG_INF = -__builtin_inff();

  // ---- layer 1 ----
  cvt_f16_kernel<<<cdiv((long long)Nn * K1, B), B, 0, stream>>>(x, ah16, (long long)Nn * K1);
  cvt_f16_T_kernel<<<cdiv((long long)K1 * FDIM, B), B, 0, stream>>>(W1, bt16, K1, FDIM);
  gemm_wmma_kernel<<<gemm_grid, B, 0, stream>>>(ah16, bt16, hbuf, Nn, K1, FDIM);
  attn_proj_kernel<<<cdiv((long long)Nn * HEADS, B), B, 0, stream>>>(
      hbuf, as1, ad1, asrc, adst, Nn);
  fill_kernel<<<cdiv((long long)Nn * HEADS, B), B, 0, stream>>>(nmax, (long long)Nn * HEADS, NEG_INF);
  fill_kernel<<<cdiv((long long)Nn * HEADS, B), B, 0, stream>>>(nsum, (long long)Nn * HEADS, 0.f);
  fill_kernel<<<cdiv((long long)Nn * FDIM,  B), B, 0, stream>>>(acc,  (long long)Nn * FDIM,  0.f);
  edge_max_kernel<<<cdiv((long long)EA * HEADS, B), B, 0, stream>>>(
      asrc, adst, esrc, edst, nmax, E, EA);
  edge_exp_kernel<<<cdiv((long long)EA * HEADS, B), B, 0, stream>>>(
      asrc, adst, esrc, edst, nmax, nsum, eexp, E, EA);
  edge_msg_kernel<<<cdiv((long long)EA * 64, B), B, 0, stream>>>(
      hbuf, esrc, edst, nsum, eexp, acc, E, EA);
  bias_elu_kernel<<<cdiv((long long)Nn * FDIM, B), B, 0, stream>>>(
      acc, b1, (long long)Nn * FDIM);

  // ---- layer 2 (acc now holds hx; gemm2 writes h2 into hbuf) ----
  cvt_f16_kernel<<<cdiv((long long)Nn * K2, B), B, 0, stream>>>(acc, ah16, (long long)Nn * K2);
  cvt_f16_T_kernel<<<cdiv((long long)K2 * FDIM, B), B, 0, stream>>>(W2, bt16, K2, FDIM);
  gemm_wmma_kernel<<<gemm_grid, B, 0, stream>>>(ah16, bt16, hbuf, Nn, K2, FDIM);
  attn_proj_kernel<<<cdiv((long long)Nn * HEADS, B), B, 0, stream>>>(
      hbuf, as2, ad2, asrc, adst, Nn);
  fill_kernel<<<cdiv((long long)Nn * HEADS, B), B, 0, stream>>>(nmax, (long long)Nn * HEADS, NEG_INF);
  fill_kernel<<<cdiv((long long)Nn * HEADS, B), B, 0, stream>>>(nsum, (long long)Nn * HEADS, 0.f);
  fill_kernel<<<cdiv((long long)Nn * FDIM,  B), B, 0, stream>>>(acc,  (long long)Nn * FDIM,  0.f);
  edge_max_kernel<<<cdiv((long long)EA * HEADS, B), B, 0, stream>>>(
      asrc, adst, esrc, edst, nmax, E, EA);
  edge_exp_kernel<<<cdiv((long long)EA * HEADS, B), B, 0, stream>>>(
      asrc, adst, esrc, edst, nmax, nsum, eexp, E, EA);
  edge_msg_kernel<<<cdiv((long long)EA * 64, B), B, 0, stream>>>(
      hbuf, esrc, edst, nsum, eexp, acc, E, EA);
  final_mean_kernel<<<cdiv((long long)Nn * CH, B), B, 0, stream>>>(
      acc, b2, (float*)d_out, Nn);
}
